// KineticFlowLayer_32658931319526
// MI455X (gfx1250) — matrix operand order repaired and checked by reading, not compiled, not run on hardware
//
#include <hip/hip_runtime.h>
#include <hip/hip_bf16.h>
#include <cstdint>
#include <cstddef>

// ---------------------------------------------------------------------------
// KineticFlowLayer for MI455X (gfx1250, wave32, WMMA bf16 + async-LDS path)
//   B=8, T=4096, D=1024
//   u = sigmoid(x@Wk^T + bk) * (x@Wv^T + bv)
//   p_t = decay * p_{t-1} + u_t            (blocked 3-phase scan)
//   vel = (p@Wq^T + bq) / mass
// ---------------------------------------------------------------------------

typedef __attribute__((ext_vector_type(16))) __bf16 v16bf;
typedef __attribute__((ext_vector_type(8)))  __bf16 v8bf;
typedef __attribute__((ext_vector_type(8)))  float  v8f;
typedef __attribute__((ext_vector_type(4)))  float  f4v;
typedef __attribute__((ext_vector_type(4)))  int    v4i;

#define D_DIM  1024
#define B_DIM  8
#define T_DIM  4096
#define M_DIM  (B_DIM * T_DIM)   // 32768 tokens
#define KB     32                // K per WMMA step
#define KSTEPS (D_DIM / KB)      // 32
#define BLK_M  128
#define BLK_N  64
#define APAD   40                // padded LDS row stride (ushorts) for A
#define WPAD   40                // padded LDS row stride (ushorts) for W
#define NCHUNK 32
#define CLEN   (T_DIM / NCHUNK)  // 128

#define AS1 __attribute__((address_space(1)))
#define AS3 __attribute__((address_space(3)))

#if defined(__has_builtin)
#  if __has_builtin(__builtin_amdgcn_global_load_async_to_lds_b128)
#    define ASYNC_OK 1
#  endif
#endif
#ifndef ASYNC_OK
#  define ASYNC_OK 0
#endif

// 16B global -> LDS copy: async DMA (ASYNCcnt) when available, else sync.
__device__ __forceinline__ void cp16_g2l(const unsigned short* g, unsigned short* l) {
#if ASYNC_OK
  __builtin_amdgcn_global_load_async_to_lds_b128(
      (AS1 v4i*)g, (AS3 v4i*)l, 0, 0);
#else
  *(uint4*)l = *(const uint4*)g;
#endif
}

__device__ __forceinline__ void wait_async_all() {
#if ASYNC_OK
#  if __has_builtin(__builtin_amdgcn_s_wait_asynccnt)
  __builtin_amdgcn_s_wait_asynccnt(0);
#  else
  asm volatile("s_wait_asynccnt 0x0" ::: "memory");
#  endif
#endif
}

__device__ __forceinline__ unsigned short f2bf(float f) {
  union { float f; unsigned u; } x; x.f = f;
  unsigned r = x.u + 0x7FFFu + ((x.u >> 16) & 1u);   // round-to-nearest-even
  return (unsigned short)(r >> 16);
}

// ------------------------------ fp32 -> bf16 -------------------------------
__global__ void cvt_f32_bf16(const float* __restrict__ src,
                             unsigned short* __restrict__ dst, int n4) {
  int i = blockIdx.x * blockDim.x + threadIdx.x;
  if (i >= n4) return;
  f4v v = ((const f4v*)src)[i];
  unsigned long long pk =
      (unsigned long long)f2bf(v[0]) |
      ((unsigned long long)f2bf(v[1]) << 16) |
      ((unsigned long long)f2bf(v[2]) << 32) |
      ((unsigned long long)f2bf(v[3]) << 48);
  ((unsigned long long*)dst)[i] = pk;
}

// --------- fused GEMM: k=A@Wk^T+bk, v=A@Wv^T+bv, u=sigmoid(k)*v ------------
__global__ __launch_bounds__(256) void gemm_kv_u(
    const unsigned short* __restrict__ Abf,   // [M, D] bf16
    const unsigned short* __restrict__ Wkbf,  // [D, D] bf16 (row e = out col)
    const unsigned short* __restrict__ Wvbf,  // [D, D] bf16
    const float* __restrict__ bk,
    const float* __restrict__ bv,
    float* __restrict__ u)                    // [M, D] fp32
{
  __shared__ unsigned short lA [2][BLK_M * APAD];
  __shared__ unsigned short lWk[2][BLK_N * WPAD];
  __shared__ unsigned short lWv[2][BLK_N * WPAD];

  const int tid  = threadIdx.x;
  const int wave = tid >> 5;
  const int lane = tid & 31;
  const int half = lane >> 4;   // 0: lanes 0-15, 1: lanes 16-31
  const int ln   = lane & 15;

  const int mBlk = blockIdx.x * BLK_M;
  const int nBlk = blockIdx.y * BLK_N;

  const int arow = tid >> 1;          // 0..127 (A stage: 2x16B per thread)
  const int aoff = (tid & 1) * 16;    // element offset within 32-wide K slab
  const int wrow = tid >> 2;          // 0..63  (W stage: 16B per thread per W)
  const int woff = (tid & 3) * 8;

  v8f accK[4], accV[4];
  const v8f vzero = {0.f,0.f,0.f,0.f,0.f,0.f,0.f,0.f};
#pragma unroll
  for (int i = 0; i < 4; ++i) { accK[i] = vzero; accV[i] = vzero; }

  auto stage = [&](int ks, int bb) {
    const int k0 = ks * KB;
    const unsigned short* gA = Abf + (size_t)(mBlk + arow) * D_DIM + k0 + aoff;
    cp16_g2l(gA,     &lA[bb][arow * APAD + aoff]);
    cp16_g2l(gA + 8, &lA[bb][arow * APAD + aoff + 8]);
    const size_t wg = (size_t)(nBlk + wrow) * D_DIM + k0 + woff;
    cp16_g2l(Wkbf + wg, &lWk[bb][wrow * WPAD + woff]);
    cp16_g2l(Wvbf + wg, &lWv[bb][wrow * WPAD + woff]);
  };

  stage(0, 0);
  for (int ks = 0; ks < KSTEPS; ++ks) {
    wait_async_all();
    __syncthreads();
    const int bb = ks & 1;
    if (ks + 1 < KSTEPS) stage(ks + 1, bb ^ 1);   // overlaps with compute below

    // A fragment (16x32 bf16): lanes 0-15 K{0..15}, lanes 16-31 K{16..31}
    v16bf afrag;
    {
      const int r = wave * 16 + ln;
      v8bf lo = *(const v8bf*)&lA[bb][r * APAD + half * 16];
      v8bf hi = *(const v8bf*)&lA[bb][r * APAD + half * 16 + 8];
      afrag = __builtin_shufflevector(lo, hi, 0,1,2,3,4,5,6,7,8,9,10,11,12,13,14,15);
    }
#pragma unroll
    for (int nf = 0; nf < 4; ++nf) {
      // B fragment (32x16 bf16): lane holds 16 contiguous K of column n
      const int n = nf * 16 + ln;
      v16bf bK = *(const v16bf*)&lWk[bb][n * WPAD + half * 16];
      accK[nf] = __builtin_amdgcn_wmma_f32_16x16x32_bf16(
          false, afrag, false, bK, (short)0, accK[nf], false, false);
      v16bf bV = *(const v16bf*)&lWv[bb][n * WPAD + half * 16];
      accV[nf] = __builtin_amdgcn_wmma_f32_16x16x32_bf16(
          false, afrag, false, bV, (short)0, accV[nf], false, false);
    }
  }

  // epilogue: u = sigmoid(k_logit) * v
  const int rbase = mBlk + wave * 16 + half * 8;
#pragma unroll
  for (int nf = 0; nf < 4; ++nf) {
    const int col = nBlk + nf * 16 + ln;
    const float bkc = bk[col];
    const float bvc = bv[col];
#pragma unroll
    for (int r = 0; r < 8; ++r) {
      float kl = accK[nf][r] + bkc;
      float vv = accV[nf][r] + bvc;
      float sg = 1.0f / (1.0f + __expf(-kl));
      u[(size_t)(rbase + r) * D_DIM + col] = sg * vv;
    }
  }
}

// --------------------- blocked scan, phase 1: local scans ------------------
__global__ __launch_bounds__(256) void scan_local(
    float* __restrict__ u, const float* __restrict__ decay,
    float* __restrict__ carry) {
  int g = blockIdx.x * 256 + threadIdx.x;          // 262144 threads
  int d = g & (D_DIM - 1);
  int rest = g >> 10;
  int c = rest & (NCHUNK - 1);
  int b = rest >> 5;
  float w = decay[d];
  float p = 0.f;
  size_t idx = ((size_t)b * T_DIM + (size_t)c * CLEN) * D_DIM + d;
#pragma unroll 4
  for (int t = 0; t < CLEN; ++t) {
    p = fmaf(p, w, u[idx]);     // p = p*decay + u_t
    u[idx] = p;                 // in-place: u becomes local p
    idx += D_DIM;
  }
  carry[((size_t)b * NCHUNK + c) * D_DIM + d] = p;
}

// ------------- phase 2: scan carries across chunks (exclusive) -------------
__global__ __launch_bounds__(256) void scan_carry(
    float* __restrict__ carry, const float* __restrict__ state_p,
    const float* __restrict__ decay) {
  int g = blockIdx.x * 256 + threadIdx.x;          // 8192 threads
  int d = g & (D_DIM - 1);
  int b = g >> 10;
  float w = decay[d];
  float wl = w;
#pragma unroll
  for (int i = 0; i < 7; ++i) wl = wl * wl;        // decay^128
  float cin = state_p[b * D_DIM + d];
  size_t base = (size_t)b * NCHUNK * D_DIM + d;
#pragma unroll
  for (int c = 0; c < NCHUNK; ++c) {
    float t = carry[base];
    carry[base] = cin;                             // exclusive prefix, in-place
    cin = fmaf(cin, wl, t);
    base += D_DIM;
  }
}

// ------- phase 3: apply carries, emit p as bf16 + final_p output -----------
__global__ __launch_bounds__(256) void scan_fix(
    const float* __restrict__ p, const float* __restrict__ carry,
    const float* __restrict__ decay, unsigned short* __restrict__ pbf,
    float* __restrict__ finp) {
  int g = blockIdx.x * 256 + threadIdx.x;
  int d = g & (D_DIM - 1);
  int rest = g >> 10;
  int c = rest & (NCHUNK - 1);
  int b = rest >> 5;
  float w = decay[d];
  float cin = carry[((size_t)b * NCHUNK + c) * D_DIM + d];
  float f = w;
  size_t idx = ((size_t)b * T_DIM + (size_t)c * CLEN) * D_DIM + d;
  float val = 0.f;
#pragma unroll 4
  for (int t = 0; t < CLEN; ++t) {
    val = fmaf(cin, f, p[idx]);   // p_true = p_local + carry_in * decay^(t+1)
    f *= w;
    pbf[idx] = f2bf(val);
    idx += D_DIM;
  }
  if (c == NCHUNK - 1) finp[b * D_DIM + d] = val;  // final_p
}

// ------------------ final GEMM: vel = (p@Wq^T + bq) / mass -----------------
__global__ __launch_bounds__(256) void gemm_q(
    const unsigned short* __restrict__ Pbf,   // [M, D] bf16
    const unsigned short* __restrict__ Wqbf,  // [D, D] bf16
    const float* __restrict__ bq,
    const float* __restrict__ mass,
    float* __restrict__ vel)                  // [M, D] fp32
{
  __shared__ unsigned short lA [2][BLK_M * APAD];
  __shared__ unsigned short lWq[2][BLK_N * WPAD];

  const int tid  = threadIdx.x;
  const int wave = tid >> 5;
  const int lane = tid & 31;
  const int half = lane >> 4;
  const int ln   = lane & 15;

  const int mBlk = blockIdx.x * BLK_M;
  const int nBlk = blockIdx.y * BLK_N;

  const int arow = tid >> 1;
  const int aoff = (tid & 1) * 16;
  const int wrow = tid >> 2;
  const int woff = (tid & 3) * 8;

  v8f acc[4];
  const v8f vzero = {0.f,0.f,0.f,0.f,0.f,0.f,0.f,0.f};
#pragma unroll
  for (int i = 0; i < 4; ++i) acc[i] = vzero;

  auto stage = [&](int ks, int bb) {
    const int k0 = ks * KB;
    const unsigned short* gA = Pbf + (size_t)(mBlk + arow) * D_DIM + k0 + aoff;
    cp16_g2l(gA,     &lA[bb][arow * APAD + aoff]);
    cp16_g2l(gA + 8, &lA[bb][arow * APAD + aoff + 8]);
    const size_t wg = (size_t)(nBlk + wrow) * D_DIM + k0 + woff;
    cp16_g2l(Wqbf + wg, &lWq[bb][wrow * WPAD + woff]);
  };

  stage(0, 0);
  for (int ks = 0; ks < KSTEPS; ++ks) {
    wait_async_all();
    __syncthreads();
    const int bb = ks & 1;
    if (ks + 1 < KSTEPS) stage(ks + 1, bb ^ 1);

    v16bf afrag;
    {
      const int r = wave * 16 + ln;
      v8bf lo = *(const v8bf*)&lA[bb][r * APAD + half * 16];
      v8bf hi = *(const v8bf*)&lA[bb][r * APAD + half * 16 + 8];
      afrag = __builtin_shufflevector(lo, hi, 0,1,2,3,4,5,6,7,8,9,10,11,12,13,14,15);
    }
#pragma unroll
    for (int nf = 0; nf < 4; ++nf) {
      const int n = nf * 16 + ln;
      v16bf bQ = *(const v16bf*)&lWq[bb][n * WPAD + half * 16];
      acc[nf] = __builtin_amdgcn_wmma_f32_16x16x32_bf16(
          false, afrag, false, bQ, (short)0, acc[nf], false, false);
    }
  }

  const int rbase = mBlk + wave * 16 + half * 8;
#pragma unroll
  for (int nf = 0; nf < 4; ++nf) {
    const int col = nBlk + nf * 16 + ln;
    const float bqc = bq[col];
    const float rm  = 1.0f / mass[col];
#pragma unroll
    for (int r = 0; r < 8; ++r) {
      vel[(size_t)(rbase + r) * D_DIM + col] = (acc[nf][r] + bqc) * rm;
    }
  }
}

// ---------------------------------------------------------------------------
extern "C" void kernel_launch(void* const* d_in, const int* in_sizes, int n_in,
                              void* d_out, int out_size, void* d_ws, size_t ws_size,
                              hipStream_t stream) {
  const float* x    = (const float*)d_in[0];   // [B,T,D]
  const float* stp  = (const float*)d_in[1];   // [B,D]
  const float* dec  = (const float*)d_in[2];   // [D]
  const float* mass = (const float*)d_in[3];   // [D]
  const float* Wq   = (const float*)d_in[4];   // [D,D]
  const float* bq   = (const float*)d_in[5];
  const float* Wk   = (const float*)d_in[6];
  const float* bk   = (const float*)d_in[7];
  const float* Wv   = (const float*)d_in[8];
  const float* bv   = (const float*)d_in[9];

  char* ws = (char*)d_ws;
  // workspace layout (p_bf16 reuses x_bf16 region once x is consumed)
  unsigned short* xbf  = (unsigned short*)(ws);                         // 64 MB
  unsigned short* wkbf = (unsigned short*)(ws + (size_t)67108864);      // 2 MB
  unsigned short* wvbf = (unsigned short*)(ws + (size_t)69206016);      // 2 MB
  unsigned short* wqbf = (unsigned short*)(ws + (size_t)71303168);      // 2 MB
  float* u             = (float*)(ws + (size_t)73400320);               // 128 MB (u -> p in-place)
  float* carry         = (float*)(ws + (size_t)73400320 + 134217728);   // 1 MB
  unsigned short* pbf  = xbf;                                           // reuse

  float* vel  = (float*)d_out;
  float* finp = vel + (size_t)M_DIM * D_DIM;

  // 1) fp32 -> bf16 conversions
  cvt_f32_bf16<<<(M_DIM * D_DIM / 4 + 255) / 256, 256, 0, stream>>>(x,  xbf,  M_DIM * D_DIM / 4);
  cvt_f32_bf16<<<(D_DIM * D_DIM / 4 + 255) / 256, 256, 0, stream>>>(Wk, wkbf, D_DIM * D_DIM / 4);
  cvt_f32_bf16<<<(D_DIM * D_DIM / 4 + 255) / 256, 256, 0, stream>>>(Wv, wvbf, D_DIM * D_DIM / 4);
  cvt_f32_bf16<<<(D_DIM * D_DIM / 4 + 255) / 256, 256, 0, stream>>>(Wq, wqbf, D_DIM * D_DIM / 4);

  // 2) fused k/v GEMM + gate -> u
  gemm_kv_u<<<dim3(M_DIM / BLK_M, D_DIM / BLK_N), 256, 0, stream>>>(
      xbf, wkbf, wvbf, bk, bv, u);

  // 3) blocked diagonal scan over T (in-place u -> p)
  scan_local<<<(B_DIM * NCHUNK * D_DIM) / 256, 256, 0, stream>>>(u, dec, carry);
  scan_carry<<<(B_DIM * D_DIM) / 256, 256, 0, stream>>>(carry, stp, dec);
  scan_fix  <<<(B_DIM * NCHUNK * D_DIM) / 256, 256, 0, stream>>>(u, carry, dec, pbf, finp);

  // 4) output GEMM -> velocities
  gemm_q<<<dim3(M_DIM / BLK_M, D_DIM / BLK_N), 256, 0, stream>>>(
      pbf, wqbf, bq, mass, vel);
}
